// VectorQuantizer_17961553232338
// MI455X (gfx1250) — matrix operand order repaired and checked by reading, not compiled, not run on hardware
//
#include <hip/hip_runtime.h>

// VQ-VAE vector quantizer for MI455X (gfx1250, wave32, WMMA).
// N=262144 rows, D=64, K=512 codes.
//
// argmin_k ||e-w_k||^2 == argmax_k (e.w_k - 0.5||w_k||^2); the -0.5||w||^2
// term is preloaded into the WMMA accumulator C, so the distance is never
// materialized. e.w_k uses bf16x3-split WMMA (a_lo*b_hi + a_hi*b_lo + a_hi*b_hi,
// f32 accumulate) -> ~f32-accurate argmin at bf16 matrix-core rate.
// Codebook fragments are staged into LDS with global_load_async_to_lds_b128
// (ASYNCcnt). loss = 1.25 * mean((q - e)^2); quantized_st forward == q.

#define N_ROWS   262144
#define D_DIM    64
#define K_CODES  512
#define ND_TOTAL (N_ROWS * D_DIM)          // 16777216
#define MAIN_BLOCKS 512                    // 8 waves * 4 rowtiles * 16 rows = 512 rows/block

typedef __attribute__((ext_vector_type(16))) __bf16 v16bf;
typedef __attribute__((ext_vector_type(8)))  float  v8f;

// ------- ws layout (uint32 units) -------
// [0,      16384)  : codebook A-fragments, hi bf16 pairs  (64 KB)
// [16384,  32768)  : codebook A-fragments, lo bf16 pairs  (64 KB)
// [32768,  33280)  : -0.5 * ||w_k||^2 (f32, 512)
// [33280,  33792)  : per-block loss partials (f32, 512)
#define WS_FRAG_HI 0
#define WS_FRAG_LO 16384
#define WS_NORMS   32768
#define WS_PART    33280
#define SMEM_DWORDS 33280                  // frags + norms in LDS (133120 B)
#define SMEM_B128   (SMEM_DWORDS / 4)      // 8320 16-byte chunks

// Pre-swizzle codebook into 16-bit A-matrix fragments (contiguous k order, not the ISA table order):
// lane l (m=l%16 = code-in-tile, h=l/16), VGPR j holds K pair at
// base = h*16 + j*2 within a 32-wide k-step.
__global__ void vq_prep_frags(const float* __restrict__ cb,
                              unsigned int* __restrict__ fragHi,
                              unsigned int* __restrict__ fragLo) {
  int tid  = blockIdx.x * blockDim.x + threadIdx.x;   // 0..16383
  int j    = tid & 7;
  int lane = (tid >> 3) & 31;
  int kk   = (tid >> 8) & 1;
  int ct   = tid >> 9;
  int m = lane & 15, h = lane >> 4;
  int c    = ct * 16 + m;
  int base = h * 16 + j * 2;
  int d    = kk * 32 + base;
  float x0 = cb[c * 64 + d];
  float x1 = cb[c * 64 + d + 1];
  __bf16 h0 = (__bf16)x0, h1 = (__bf16)x1;
  __bf16 l0 = (__bf16)(x0 - (float)h0), l1 = (__bf16)(x1 - (float)h1);
  union { __bf16 b[2]; unsigned int u; } ph, pl;
  ph.b[0] = h0; ph.b[1] = h1;
  pl.b[0] = l0; pl.b[1] = l1;
  fragHi[tid] = ph.u;
  fragLo[tid] = pl.u;
}

__global__ void vq_prep_norms(const float* __restrict__ cb, float* __restrict__ cn) {
  int k = blockIdx.x * blockDim.x + threadIdx.x;
  if (k < K_CODES) {
    float s = 0.f;
    #pragma unroll 8
    for (int d = 0; d < 64; ++d) { float v = cb[k * 64 + d]; s += v * v; }
    cn[k] = -0.5f * s;                     // folded into WMMA accumulator init
  }
}

__global__ void __launch_bounds__(256)
vq_main(const float* __restrict__ e, const float* __restrict__ cb,
        const unsigned int* __restrict__ wsu, float* __restrict__ partials,
        float* __restrict__ out) {
  extern __shared__ unsigned int smem[];

  // Stage pre-swizzled codebook (hi+lo) + folded norms into LDS once per
  // workgroup using async memory->LDS DMA (no VGPR round-trip, ASYNCcnt).
  {
    const unsigned lds_base = (unsigned)(uintptr_t)smem;
    for (int i = threadIdx.x; i < SMEM_B128; i += 256) {
      unsigned goff = (unsigned)i * 16u;
      unsigned loff = lds_base + goff;
      asm volatile("global_load_async_to_lds_b128 %0, %1, %2"
                   :: "v"(loff), "v"(goff), "s"(wsu)
                   : "memory");
    }
    asm volatile("s_wait_asynccnt 0" ::: "memory");
  }
  __syncthreads();

  const unsigned int* sHi = smem;
  const unsigned int* sLo = smem + WS_FRAG_LO;
  const float*        scn = (const float*)(smem + WS_NORMS);

  const int wave = threadIdx.x >> 5;
  const int lane = threadIdx.x & 31;
  const int n = lane & 15;          // e-row within tile (B/D column)
  const int h = lane >> 4;          // lane half
  float lsum = 0.f;
  __shared__ float wsum[8];

  for (int rt = 0; rt < 4; ++rt) {
    const int rowTile = blockIdx.x * 32 + wave * 4 + rt;
    const int row = rowTile * 16 + n;
    const float4* e4p = (const float4*)(e + (size_t)row * 64);

    // Each lane keeps 32 f32 e-values: dims [h*16,h*16+16) of each 32-wide k-step.
    float4 e4[8];
    #pragma unroll
    for (int kk = 0; kk < 2; ++kk)
      #pragma unroll
      for (int q = 0; q < 4; ++q)
        e4[kk * 4 + q] = e4p[kk * 8 + h * 4 + q];

    // Build B fragments (32x16 bf16, contiguous k order matching A: lanes 0-15 hold K 0-15 pairs,
    // lanes 16-31 hold K 16-31) with hi/lo split.
    v16bf bhi[2], blo[2];
    const float* ef = (const float*)e4;
    #pragma unroll
    for (int kk = 0; kk < 2; ++kk)
      #pragma unroll
      for (int t = 0; t < 16; ++t) {
        float x = ef[kk * 16 + t];
        __bf16 hb = (__bf16)x;
        __bf16 lb = (__bf16)(x - (float)hb);
        bhi[kk][t] = hb;
        blo[kk][t] = lb;
      }

    float lmax = -3.4e38f;          // max of (e.w - 0.5||w||^2)  == argmin dist
    int   limax = 0;
    for (int ct = 0; ct < 32; ++ct) {
      v16bf aH0 = *(const v16bf*)(sHi + (ct * 2 + 0) * 256 + lane * 8);
      v16bf aH1 = *(const v16bf*)(sHi + (ct * 2 + 1) * 256 + lane * 8);
      v16bf aL0 = *(const v16bf*)(sLo + (ct * 2 + 0) * 256 + lane * 8);
      v16bf aL1 = *(const v16bf*)(sLo + (ct * 2 + 1) * 256 + lane * 8);
      // Accumulator seeded with -0.5||w||^2 for this lane's 8 codes (32B v8f load).
      v8f acc = *(const v8f*)(scn + ct * 16 + h * 8);
      // bf16x3: small cross terms first, then hi*hi.
      acc = __builtin_amdgcn_wmma_f32_16x16x32_bf16(false, aL0, false, bhi[0], (short)0, acc, false, false);
      acc = __builtin_amdgcn_wmma_f32_16x16x32_bf16(false, aL1, false, bhi[1], (short)0, acc, false, false);
      acc = __builtin_amdgcn_wmma_f32_16x16x32_bf16(false, aH0, false, blo[0], (short)0, acc, false, false);
      acc = __builtin_amdgcn_wmma_f32_16x16x32_bf16(false, aH1, false, blo[1], (short)0, acc, false, false);
      acc = __builtin_amdgcn_wmma_f32_16x16x32_bf16(false, aH0, false, bhi[0], (short)0, acc, false, false);
      acc = __builtin_amdgcn_wmma_f32_16x16x32_bf16(false, aH1, false, bhi[1], (short)0, acc, false, false);
      // D tile: VGPR v at lane l -> code = ct*16 + h*8 + v, column n = l%16.
      #pragma unroll
      for (int v = 0; v < 8; ++v) {
        float sc = acc[v];
        int code = ct * 16 + h * 8 + v;
        if (sc > lmax) { lmax = sc; limax = code; }  // strict > -> first index on ties
      }
    }
    // Combine the two lane halves; prefer lower code index on ties (argmin semantics).
    float ov = __shfl_xor(lmax, 16);
    int   oi = __shfl_xor(limax, 16);
    if (ov > lmax || (ov == lmax && oi < limax)) { lmax = ov; limax = oi; }

    // Gather the selected codebook row (f32, exact) and emit quantized + loss.
    const float4* w4 = (const float4*)(cb + (size_t)limax * 64);
    float4* o4 = (float4*)(out + (size_t)row * 64);
    #pragma unroll
    for (int kk = 0; kk < 2; ++kk)
      #pragma unroll
      for (int q = 0; q < 4; ++q) {
        float4 w = w4[kk * 8 + h * 4 + q];
        float4 ev = e4[kk * 4 + q];
        o4[kk * 8 + h * 4 + q] = w;
        float dx = w.x - ev.x, dy = w.y - ev.y, dz = w.z - ev.z, dw = w.w - ev.w;
        lsum += dx * dx + dy * dy + dz * dz + dw * dw;
      }
  }

  // Deterministic fixed-order reduction: shuffle tree -> per-wave -> block partial.
  #pragma unroll
  for (int off = 16; off > 0; off >>= 1) lsum += __shfl_down(lsum, off);
  if (lane == 0) wsum[wave] = lsum;
  __syncthreads();
  if (threadIdx.x == 0) {
    float t = 0.f;
    for (int i = 0; i < 8; ++i) t += wsum[i];
    partials[blockIdx.x] = t;
  }
}

__global__ void vq_loss(const float* __restrict__ partials, float* __restrict__ out) {
  if (blockIdx.x == 0 && threadIdx.x == 0) {
    float t = 0.f;
    for (int i = 0; i < MAIN_BLOCKS; ++i) t += partials[i];   // fixed order: deterministic
    out[ND_TOTAL] = 1.25f * t / (float)ND_TOTAL;              // q_loss + 0.25*e_loss
  }
}

extern "C" void kernel_launch(void* const* d_in, const int* in_sizes, int n_in,
                              void* d_out, int out_size, void* d_ws, size_t ws_size,
                              hipStream_t stream) {
  const float* e  = (const float*)d_in[1];   // d_in[0] = x (unused by reference math)
  const float* cb = (const float*)d_in[2];
  float* out = (float*)d_out;

  unsigned int* ws = (unsigned int*)d_ws;
  unsigned int* fragHi = ws + WS_FRAG_HI;
  unsigned int* fragLo = ws + WS_FRAG_LO;
  float* cnorm    = (float*)(ws + WS_NORMS);
  float* partials = (float*)(ws + WS_PART);

  // Allow >64KB dynamic LDS (130KB codebook stage); host-side call, not captured.
  static_cast<void>(hipFuncSetAttribute((const void*)vq_main,
                                        hipFuncAttributeMaxDynamicSharedMemorySize,
                                        SMEM_DWORDS * 4));

  vq_prep_frags<<<64, 256, 0, stream>>>(cb, fragHi, fragLo);
  vq_prep_norms<<<2, 256, 0, stream>>>(cb, cnorm);
  vq_main<<<MAIN_BLOCKS, 256, SMEM_DWORDS * 4, stream>>>(e, cb, ws, partials, out);
  vq_loss<<<1, 64, 0, stream>>>(partials, out);
}
